// FSUConv2d_25469156065828
// MI455X (gfx1250) — 8-bit matrix operand order repaired and checked by reading, compiled, not run on hardware
//
#include <hip/hip_runtime.h>
#include <hip/hip_bf16.h>

typedef __attribute__((ext_vector_type(8))) int v8i;
typedef __attribute__((ext_vector_type(2))) int v2i;

#define N_IMG   16
#define CIN     64
#define HH      56
#define WW      56
#define COUT    64
#define KK      3
#define HP      58
#define WP      58
#define L_PIX   (HH * WW)
#define ROW_TILES_PER_IMG (L_PIX / 16)
#define TOTAL_ROW_TILES (N_IMG * ROW_TILES_PER_IMG)
#define LDS_STRIDE 20

__global__ __launch_bounds__(256) void fsu_prep_xt(
        const float* __restrict__ x, unsigned char* __restrict__ xpad) {
    __shared__ unsigned char tile[CIN][WW + 1];
    const int t = threadIdx.x;
    const int h = blockIdx.x % HH;
    const int n = blockIdx.x / HH;

    #pragma unroll
    for (int rep = 0; rep < 14; ++rep) {
        int idx = rep * 256 + t;
        int c = idx / WW;
        int w = idx % WW;
        float v = x[(((size_t)n * CIN + c) * HH + h) * WW + w];
        tile[c][w] = (v != 0.0f) ? 1 : 0;
    }
    __syncthreads();

    #pragma unroll
    for (int rep = 0; rep < 14; ++rep) {
        int idx = rep * 256 + t;
        int w = idx / CIN;
        int c = idx % CIN;
        xpad[(((size_t)n * HP + (h + 1)) * WP + (w + 1)) * CIN + c] = tile[c][w];
    }
}

__global__ void fsu_prep_w(const float* __restrict__ weight,
                           int* __restrict__ bpack) {
    int idx = blockIdx.x * blockDim.x + threadIdx.x;
    if (idx >= 9 * 4 * 32 * 8) return;
    int j    = idx & 7;
    int lane = (idx >> 3) & 31;
    int tt   = (idx >> 8) & 3;
    int kk   = idx >> 10;
    int halfB = lane >> 4;
    int col   = lane & 15;
    int o     = tt * 16 + col;
    int kh = kk / 3, kw = kk % 3;
    int Kbase = (j >> 1) * 16 + halfB * 8 + (j & 1) * 4;
    unsigned int dw = 0;
    #pragma unroll
    for (int b = 0; b < 4; ++b) {
        int c = Kbase + b;
        float wv = weight[(((size_t)o * CIN + c) * KK + kh) * KK + kw];
        dw |= ((wv > 0.0f) ? 1u : 0u) << (8 * b);
    }
    bpack[idx] = (int)dw;
}

__global__ void fsu_chansum(const unsigned char* __restrict__ xpad,
                            int* __restrict__ chansum, int total) {
    int idx = blockIdx.x * blockDim.x + threadIdx.x;
    if (idx >= total) return;
    const unsigned int* p = (const unsigned int*)(xpad + (size_t)idx * CIN);
    int s = 0;
    #pragma unroll
    for (int i = 0; i < 16; ++i) s += __builtin_popcount(p[i]);
    chansum[idx] = s;
}

__global__ void fsu_rowsum(const int* __restrict__ chansum,
                           int* __restrict__ rowsum, int total) {
    int idx = blockIdx.x * blockDim.x + threadIdx.x;
    if (idx >= total) return;
    int l = idx % L_PIX;
    int n = idx / L_PIX;
    int oh = l / WW, ow = l % WW;
    int s = 0;
    #pragma unroll
    for (int kh = 0; kh < 3; ++kh)
        #pragma unroll
        for (int kw = 0; kw < 3; ++kw)
            s += chansum[((size_t)n * HP + (oh + kh)) * WP + (ow + kw)];
    rowsum[idx] = s;
}

__global__ void fsu_prep_thresh(const float* __restrict__ weight,
                                const float* __restrict__ bias,
                                int* __restrict__ thresh) {
    int o = blockIdx.x * blockDim.x + threadIdx.x;
    if (o >= COUT) return;
    int cs = 0;
    for (int k = 0; k < CIN * KK * KK; ++k)
        cs += (weight[(size_t)o * CIN * KK * KK + k] > 0.0f) ? 1 : 0;
    int bb = (bias[o] > 0.0f) ? 1 : 0;
    thresh[o] = 1 + cs - bb;
}

__global__ __launch_bounds__(256) void fsu_gemm(
        const unsigned char* __restrict__ xpad,
        const int* __restrict__ bpack,
        const int* __restrict__ rowsum,
        const int* __restrict__ thresh,
        float* __restrict__ out) {
    __shared__ float lds[8][COUT * LDS_STRIDE];

    const int lane = threadIdx.x & 31;
    const int wave = threadIdx.x >> 5;
    const int tile = blockIdx.x * 8 + wave;
    const int n     = tile / ROW_TILES_PER_IMG;
    const int lbase = (tile % ROW_TILES_PER_IMG) * 16;

    const int half = lane >> 4;
    const int r    = lane & 15;
    const int l    = lbase + r;
    const int oh   = l / WW;
    const int ow   = l % WW;

    const unsigned char* rowbase =
        xpad + (((size_t)n * HP + oh) * WP + ow) * CIN + half * 8;

    v8i acc[4];
    #pragma unroll
    for (int t = 0; t < 4; ++t) acc[t] = (v8i)0;

    #pragma unroll
    for (int kk = 0; kk < 9; ++kk) {
        const int kh = kk / 3, kw = kk % 3;
        const unsigned char* p = rowbase + ((size_t)kh * WP + kw) * CIN;

        v2i q0 = *(const v2i*)(p);
        v2i q1 = *(const v2i*)(p + 16);
        v2i q2 = *(const v2i*)(p + 32);
        v2i q3 = *(const v2i*)(p + 48);
        v8i a;
        a[0] = q0.x; a[1] = q0.y;
        a[2] = q1.x; a[3] = q1.y;
        a[4] = q2.x; a[5] = q2.y;
        a[6] = q3.x; a[7] = q3.y;

        #pragma unroll
        for (int tt = 0; tt < 4; ++tt) {
            const int* bp = bpack + (((size_t)(kk * 4 + tt)) * 32 + lane) * 8;
            v8i b = *(const v8i*)bp;
            acc[tt] = __builtin_amdgcn_wmma_i32_16x16x64_iu8(
                false, a, false, b, acc[tt], false, false);
        }
    }

    int rs[8];
    #pragma unroll
    for (int j = 0; j < 8; ++j)
        rs[j] = rowsum[(size_t)n * L_PIX + lbase + j + half * 8];

    float* ldsw = &lds[wave][0];
    #pragma unroll
    for (int tt = 0; tt < 4; ++tt) {
        const int o  = tt * 16 + r;
        const int th = thresh[o];
        #pragma unroll
        for (int j = 0; j < 8; ++j) {
            const int m = j + half * 8;
            ldsw[o * LDS_STRIDE + m] =
                (2 * acc[tt][j] - rs[j] >= th) ? 1.0f : 0.0f;
        }
    }

    #pragma unroll
    for (int it = 0; it < 8; ++it) {
        const int idx = it * 32 + lane;
        const int o   = idx >> 2;
        const int g   = idx & 3;
        float4 v = *(const float4*)&ldsw[o * LDS_STRIDE + g * 4];
        *(float4*)(out + ((size_t)n * COUT + o) * L_PIX + lbase + g * 4) = v;
    }
}

extern "C" void kernel_launch(void* const* d_in, const int* in_sizes, int n_in,
                              void* d_out, int out_size, void* d_ws, size_t ws_size,
                              hipStream_t stream) {
    const float* x      = (const float*)d_in[0];
    const float* weight = (const float*)d_in[1];
    const float* bias   = (const float*)d_in[2];
    float* out          = (float*)d_out;

    char* ws = (char*)d_ws;
    unsigned char* xpad = (unsigned char*)ws;
    size_t off = (size_t)N_IMG * HP * WP * CIN;
    int* bpack = (int*)(ws + off);            off += 9 * 4 * 32 * 8 * 4;
    int* chansum = (int*)(ws + off);          off += (size_t)N_IMG * HP * WP * 4;
    int* rowsum  = (int*)(ws + off);          off += (size_t)N_IMG * L_PIX * 4;
    int* thresh  = (int*)(ws + off);

    hipMemsetAsync(xpad, 0, (size_t)N_IMG * HP * WP * CIN, stream);

    fsu_prep_xt<<<N_IMG * HH, 256, 0, stream>>>(x, xpad);

    const int totalw = 9 * 4 * 32 * 8;
    fsu_prep_w<<<(totalw + 255) / 256, 256, 0, stream>>>(weight, bpack);

    const int totalcs = N_IMG * HP * WP;
    fsu_chansum<<<(totalcs + 255) / 256, 256, 0, stream>>>(xpad, chansum, totalcs);

    const int totalrs = N_IMG * L_PIX;
    fsu_rowsum<<<(totalrs + 255) / 256, 256, 0, stream>>>(chansum, rowsum, totalrs);

    fsu_prep_thresh<<<1, 64, 0, stream>>>(weight, bias, thresh);

    fsu_gemm<<<TOTAL_ROW_TILES / 8, 256, 0, stream>>>(xpad, bpack, rowsum, thresh, out);
}
